// TemporalInvertedMultiHeadAttention_84318797955259
// MI455X (gfx1250) — matrix operand order repaired and checked by reading, not compiled, not run on hardware
//
#include <hip/hip_runtime.h>
#include <hip/hip_bf16.h>

// ---------------------------------------------------------------------------
// TemporalInvertedMultiHeadAttention for MI455X (gfx1250, wave32, WMMA)
// ---------------------------------------------------------------------------

typedef __attribute__((ext_vector_type(16))) _Float16 v16h;
typedef __attribute__((ext_vector_type(8)))  _Float16 v8h;
typedef __attribute__((ext_vector_type(8)))  float    v8f;

static constexpr int Bz  = 8;          // batch
static constexpr int Tz  = 6;          // time
static constexpr int Nz  = 24;         // queries per time
static constexpr int Lz  = 1024;       // keys per time
static constexpr int Dz  = 512;        // model dim
static constexpr int Hz  = 8;          // heads
static constexpr int DHz = 64;         // head dim
static constexpr int TN  = Tz * Nz;    // 144
static constexpr int TL  = Tz * Lz;    // 6144
static constexpr float EPSF = 1e-5f;

// ---------------------------------------------------------------------------
// f32 -> f16 convert (with scale folding; used for weights)
// ---------------------------------------------------------------------------
__global__ void cvt_f32_f16_kernel(const float* __restrict__ src,
                                   _Float16* __restrict__ dst,
                                   int n, float scale) {
    int i = blockIdx.x * blockDim.x + threadIdx.x;
    if (i < n) dst[i] = (_Float16)(src[i] * scale);
}

// ---------------------------------------------------------------------------
// Generic GEMM: Y(M x 512) = X(M x 512) @ W(512 x 512)^T
//   B-operand fragment element [k][n] = W[n][k] -> k-contiguous in memory.
//   128 threads = 4 waves; block tile 16 rows x 64 cols; wave tile 16x16.
// ---------------------------------------------------------------------------
template <typename InT, typename OutT>
__global__ void gemm_xw_kernel(const InT* __restrict__ X,
                               const _Float16* __restrict__ Wh,
                               OutT* __restrict__ Y) {
    const int K    = Dz;
    const int wave = threadIdx.x >> 5;
    const int lane = threadIdx.x & 31;
    const int row0 = blockIdx.y * 16;
    const int col0 = blockIdx.x * 64 + wave * 16;

    const int rowA = row0 + (lane & 15);
    const int kofA = (lane & 16) ? 16 : 0;    // A: K 0..15 vs 16..31
    const int colB = col0 + (lane & 15);
    const int kofB = (lane & 16) ? 16 : 0;    // B: K 0..15 vs 16..31

    const InT* Abase = X + (size_t)rowA * K;
    const _Float16* Bbase = Wh + (size_t)colB * K;

    v8f acc = {};
    for (int k0 = 0; k0 < K; k0 += 32) {
        const InT* pa = Abase + k0 + kofA;
        v16h af;
#pragma unroll
        for (int j = 0; j < 8; ++j) {
            af[j]     = (_Float16)pa[j];
            af[8 + j] = (_Float16)pa[8 + j];
        }
        v16h bf = *(const v16h*)(Bbase + k0 + kofB);   // 32B contiguous
        acc = __builtin_amdgcn_wmma_f32_16x16x32_f16(
            false, af, false, bf, (short)0, acc, false, false);
    }

    const int rof = (lane & 16) ? 8 : 0;
#pragma unroll
    for (int i = 0; i < 8; ++i)
        Y[(size_t)(row0 + rof + i) * Dz + col0 + (lane & 15)] = (OutT)acc[i];
}

// ---------------------------------------------------------------------------
// V projection with head-transposed store:
//   vpt[b][d][m] = (v @ Wv^T)[b*TL + m][d]   layout (B, D, TL), m-contiguous
// 16x64 output tile is transposed through LDS so global stores are 16B-chunk
// coalesced instead of 2-byte scatters.
// ---------------------------------------------------------------------------
__global__ void gemm_vw_transposed_kernel(const float* __restrict__ X,
                                          const _Float16* __restrict__ Wh,
                                          _Float16* __restrict__ Yt) {
    const int K    = Dz;
    const int wave = threadIdx.x >> 5;
    const int lane = threadIdx.x & 31;
    const int row0 = blockIdx.y * 16;            // global m-row (within B*TL)
    const int dcol0 = blockIdx.x * 64;           // d-tile base
    const int col0 = dcol0 + wave * 16;

    const int rowA = row0 + (lane & 15);
    const int kofA = (lane & 16) ? 16 : 0;
    const int colB = col0 + (lane & 15);
    const int kofB = (lane & 16) ? 16 : 0;

    const float* Abase = X + (size_t)rowA * K;
    const _Float16* Bbase = Wh + (size_t)colB * K;

    v8f acc = {};
    for (int k0 = 0; k0 < K; k0 += 32) {
        const float* pa = Abase + k0 + kofA;
        v16h af;
#pragma unroll
        for (int j = 0; j < 8; ++j) {
            af[j]     = (_Float16)pa[j];
            af[8 + j] = (_Float16)pa[8 + j];
        }
        v16h bf = *(const v16h*)(Bbase + k0 + kofB);
        acc = __builtin_amdgcn_wmma_f32_16x16x32_f16(
            false, af, false, bf, (short)0, acc, false, false);
    }

    // Transpose 16(m) x 64(d) tile through LDS: tile[d][m]
    __shared__ _Float16 tile[64][16];
    const int rof  = (lane & 16) ? 8 : 0;
    const int dloc = wave * 16 + (lane & 15);
#pragma unroll
    for (int i = 0; i < 8; ++i)
        tile[dloc][rof + i] = (_Float16)acc[i];
    __syncthreads();

    // Coalesced store: each thread writes 8 contiguous halves of one d-row.
    const int d  = threadIdx.x >> 1;             // 0..63
    const int c  = (threadIdx.x & 1) * 8;        // 0 or 8
    const int bb = row0 / TL;                    // batch (tile never spans b)
    const int m0 = row0 - bb * TL;
    _Float16* dst = Yt + ((size_t)bb * Dz + dcol0 + d) * TL + m0 + c;
    *(v8h*)dst = *(const v8h*)&tile[d][c];
}

// ---------------------------------------------------------------------------
// Scores: per (b,h), S(144 x 6144) = qp(b,h) @ kp(b,h)^T, masked -> -1e30
//   Writes f32 scores into the attn_vis region of d_out (scratch reuse).
// ---------------------------------------------------------------------------
__global__ void qk_scores_kernel(const _Float16* __restrict__ qph,
                                 const _Float16* __restrict__ kph,
                                 const unsigned char* __restrict__ mask,
                                 float* __restrict__ scores) {
    const int bh   = blockIdx.z;
    const int h    = bh % Hz;
    const int b    = bh / Hz;
    const int wave = threadIdx.x >> 5;
    const int lane = threadIdx.x & 31;
    const int row0 = blockIdx.y * 16;
    const int col0 = blockIdx.x * 64 + wave * 16;

    const int rowA = row0 + (lane & 15);
    const int kofA = (lane & 16) ? 16 : 0;
    const int colB = col0 + (lane & 15);
    const int kofB = (lane & 16) ? 16 : 0;

    const _Float16* Abase = qph + ((size_t)b * TN + rowA) * Dz + h * DHz;
    const _Float16* Bbase = kph + ((size_t)b * TL + colB) * Dz + h * DHz;

    v8f acc = {};
#pragma unroll
    for (int k0 = 0; k0 < DHz; k0 += 32) {
        v8h alo = *(const v8h*)(Abase + k0 + kofA);
        v8h ahi = *(const v8h*)(Abase + k0 + kofA + 8);
        v16h af;
#pragma unroll
        for (int j = 0; j < 8; ++j) { af[j] = alo[j]; af[8 + j] = ahi[j]; }
        v16h bf = *(const v16h*)(Bbase + k0 + kofB);
        acc = __builtin_amdgcn_wmma_f32_16x16x32_f16(
            false, af, false, bf, (short)0, acc, false, false);
    }

    const int rof = (lane & 16) ? 8 : 0;
    const int col = col0 + (lane & 15);
    float* out = scores + (size_t)bh * TN * TL;
#pragma unroll
    for (int i = 0; i < 8; ++i) {
        const int row = row0 + rof + i;
        float s = acc[i];
        if (mask[(size_t)row * TL + col]) s = -1.0e30f;  // finite -inf substitute
        out[(size_t)row * TL + col] = s;
    }
}

// ---------------------------------------------------------------------------
// Cross-head softmax stats: for each (b,t,tl), online max / sum-exp over the
// 192 values { scores[b,h,t*N+n,tl] : h<8, n<24 }. Single pass over 226 MB.
// ---------------------------------------------------------------------------
__global__ void group_stats_kernel(const float* __restrict__ scores,
                                   float* __restrict__ gmax,
                                   float* __restrict__ gsum) {
    const int tl = blockIdx.x * blockDim.x + threadIdx.x;     // 0..TL-1
    const int bt = blockIdx.y;                                // 0..B*T-1
    const int b = bt / Tz, t = bt % Tz;
    float m = -1.0e30f, s = 0.0f;
#pragma unroll 1
    for (int h = 0; h < Hz; ++h) {
        const float* p = scores + (((size_t)(b * Hz + h) * TN + t * Nz) * TL) + tl;
#pragma unroll 4
        for (int n = 0; n < Nz; ++n) {
            float x  = p[(size_t)n * TL];
            float nm = fmaxf(m, x);
            s = s * __expf(m - nm) + __expf(x - nm);
            m = nm;
        }
    }
    gmax[(size_t)bt * TL + tl] = m;
    gsum[(size_t)bt * TL + tl] = s;
}

// ---------------------------------------------------------------------------
// In-place scores -> a = exp(x-max)/sumexp (masked -> 0), plus per-row sum
// over the key axis (LDS block reduction). One block per (b,h,row). float4.
// ---------------------------------------------------------------------------
__global__ void softmax_renorm_kernel(float* __restrict__ a,
                                      const float* __restrict__ gmax,
                                      const float* __restrict__ gsum,
                                      const unsigned char* __restrict__ mask,
                                      float* __restrict__ rowsum) {
    const int bhr = blockIdx.x;            // 0..B*H*TN-1
    const int row = bhr % TN;
    const int bh  = bhr / TN;
    const int b   = bh / Hz;
    const int t   = row / Nz;

    float4* ap4 = (float4*)(a + ((size_t)bh * TN + row) * TL);
    const float4* mx4 = (const float4*)(gmax + (size_t)(b * Tz + t) * TL);
    const float4* sm4 = (const float4*)(gsum + (size_t)(b * Tz + t) * TL);
    const uchar4* mk4 = (const uchar4*)(mask + (size_t)row * TL);

    float local = 0.0f;
#pragma unroll 2
    for (int i = threadIdx.x; i < TL / 4; i += blockDim.x) {
        float4 x = ap4[i];
        float4 mxv = mx4[i];
        float4 smv = sm4[i];
        uchar4 mv = mk4[i];
        float4 r;
        r.x = mv.x ? 0.0f : (__expf(x.x - mxv.x) / smv.x);
        r.y = mv.y ? 0.0f : (__expf(x.y - mxv.y) / smv.y);
        r.z = mv.z ? 0.0f : (__expf(x.z - mxv.z) / smv.z);
        r.w = mv.w ? 0.0f : (__expf(x.w - mxv.w) / smv.w);
        ap4[i] = r;                // final attn_vis value
        local += r.x + r.y + r.z + r.w;
    }

    __shared__ float red[256];
    red[threadIdx.x] = local;
    __syncthreads();
    for (int s = 128; s > 0; s >>= 1) {
        if (threadIdx.x < (unsigned)s) red[threadIdx.x] += red[threadIdx.x + s];
        __syncthreads();
    }
    if (threadIdx.x == 0) rowsum[bhr] = red[0];
}

// ---------------------------------------------------------------------------
// AV GEMM per (b,h): outp[b,row,h*DH+d] = sum_m (a*inv)[row,m] * vpt[b,d,m].
//   A loaded f32, scaled by 1/(rowsum+eps) in registers, converted to f16.
//   B-fragment element [k=m][col=d] = vpt[b][h*DH+d][m] -> single 32B load.
// ---------------------------------------------------------------------------
__global__ void av_gemm_kernel(const float* __restrict__ a,
                               const float* __restrict__ rowsum,
                               const _Float16* __restrict__ vpt,
                               _Float16* __restrict__ outp) {
    const int bh   = blockIdx.z;
    const int h    = bh % Hz;
    const int b    = bh / Hz;
    const int wave = threadIdx.x >> 5;
    const int lane = threadIdx.x & 31;
    const int row0 = blockIdx.y * 16;
    const int col0 = wave * 16;                 // within DH (4 waves x 16 = 64)

    const int rowA = row0 + (lane & 15);
    const int kofA = (lane & 16) ? 16 : 0;
    const int colB = col0 + (lane & 15);
    const int kofB = (lane & 16) ? 16 : 0;

    const float inv = 1.0f / (rowsum[(size_t)bh * TN + rowA] + EPSF);

    const float* Abase = a + ((size_t)bh * TN + rowA) * TL;
    const _Float16* Bbase = vpt + ((size_t)b * Dz + h * DHz + colB) * TL;

    v8f acc = {};
    for (int k0 = 0; k0 < TL; k0 += 32) {
        const float* pa = Abase + k0 + kofA;
        v16h af;
#pragma unroll
        for (int j = 0; j < 8; ++j) {
            af[j]     = (_Float16)(pa[j]      * inv);
            af[8 + j] = (_Float16)(pa[8 + j]  * inv);
        }
        v16h bf = *(const v16h*)(Bbase + k0 + kofB);   // 32B contiguous (m-major)
        acc = __builtin_amdgcn_wmma_f32_16x16x32_f16(
            false, af, false, bf, (short)0, acc, false, false);
    }

    const int rof = (lane & 16) ? 8 : 0;
    _Float16* op = outp + (size_t)b * TN * Dz + h * DHz;
#pragma unroll
    for (int i = 0; i < 8; ++i)
        op[(size_t)(row0 + rof + i) * Dz + col0 + (lane & 15)] = (_Float16)acc[i];
}

// ---------------------------------------------------------------------------
// Host-side launch
// ---------------------------------------------------------------------------
extern "C" void kernel_launch(void* const* d_in, const int* in_sizes, int n_in,
                              void* d_out, int out_size, void* d_ws, size_t ws_size,
                              hipStream_t stream) {
    const float* q  = (const float*)d_in[0];     // (B,T,N,D) = (B,TN,D)
    const float* k  = (const float*)d_in[1];     // (B,T,L,D) = (B,TL,D)
    const float* v  = (const float*)d_in[2];
    const unsigned char* mask = (const unsigned char*)d_in[3];  // (TN,TL) bool
    const float* Wq = (const float*)d_in[4];
    const float* Wk = (const float*)d_in[5];
    const float* Wv = (const float*)d_in[6];
    const float* Wo = (const float*)d_in[7];

    float* out  = (float*)d_out;                         // (B,T,N,D)
    float* avis = out + (size_t)Bz * TN * Dz;            // (B,H,TN,TL); also scores scratch

    // Workspace carve-up (~108 MB)
    char* ws = (char*)d_ws;
    auto take = [&](size_t bytes) { char* p = ws; ws += (bytes + 255) & ~(size_t)255; return p; };
    _Float16* Wqh  = (_Float16*)take((size_t)Dz * Dz * 2);
    _Float16* Wkh  = (_Float16*)take((size_t)Dz * Dz * 2);
    _Float16* Wvh  = (_Float16*)take((size_t)Dz * Dz * 2);
    _Float16* Woh  = (_Float16*)take((size_t)Dz * Dz * 2);
    _Float16* qph  = (_Float16*)take((size_t)Bz * TN * Dz * 2);
    _Float16* kph  = (_Float16*)take((size_t)Bz * TL * Dz * 2);
    _Float16* vpt  = (_Float16*)take((size_t)Bz * Dz * TL * 2);   // (B, D, TL)
    _Float16* outp = (_Float16*)take((size_t)Bz * TN * Dz * 2);
    float* gmax    = (float*)take((size_t)Bz * Tz * TL * 4);
    float* gsum    = (float*)take((size_t)Bz * Tz * TL * 4);
    float* rowsum  = (float*)take((size_t)Bz * Hz * TN * 4);

    const int WN = Dz * Dz;
    const float qscale = 0.125f;   // DH^-0.5 folded into Wq (projection is linear)
    cvt_f32_f16_kernel<<<(WN + 255) / 256, 256, 0, stream>>>(Wq, Wqh, WN, qscale);
    cvt_f32_f16_kernel<<<(WN + 255) / 256, 256, 0, stream>>>(Wk, Wkh, WN, 1.0f);
    cvt_f32_f16_kernel<<<(WN + 255) / 256, 256, 0, stream>>>(Wv, Wvh, WN, 1.0f);
    cvt_f32_f16_kernel<<<(WN + 255) / 256, 256, 0, stream>>>(Wo, Woh, WN, 1.0f);

    // Projections (WMMA f16 -> f32 acc -> f16 result)
    gemm_xw_kernel<float, _Float16>
        <<<dim3(Dz / 64, (Bz * TN) / 16), 128, 0, stream>>>(q, Wqh, qph);
    gemm_xw_kernel<float, _Float16>
        <<<dim3(Dz / 64, (Bz * TL) / 16), 128, 0, stream>>>(k, Wkh, kph);
    // V projection stored head-transposed (B, D, TL) via LDS transpose
    gemm_vw_transposed_kernel
        <<<dim3(Dz / 64, (Bz * TL) / 16), 128, 0, stream>>>(v, Wvh, vpt);

    // Scores + mask, written into attn_vis region
    qk_scores_kernel<<<dim3(TL / 64, TN / 16, Bz * Hz), 128, 0, stream>>>(
        qph, kph, mask, avis);

    // Cross-head online softmax stats over (H*N)=192 per (b,t,tl)
    group_stats_kernel<<<dim3(TL / 256, Bz * Tz), 256, 0, stream>>>(avis, gmax, gsum);

    // In-place exp/normalize (final attn_vis) + per-row sums for renorm
    softmax_renorm_kernel<<<Bz * Hz * TN, 256, 0, stream>>>(avis, gmax, gsum, mask, rowsum);

    // AV (renorm by 1/(rowsum+eps) fused into A-operand load)
    av_gemm_kernel<<<dim3(1, TN / 16, Bz * Hz), 128, 0, stream>>>(
        avis, rowsum, vpt, outp);

    // Output projection -> fp32 out
    gemm_xw_kernel<_Float16, float>
        <<<dim3(Dz / 64, (Bz * TN) / 16), 128, 0, stream>>>(outp, Woh, out);
}
